// HeteroGNNs_57509612094235
// MI455X (gfx1250) — compile-verified, matrix operand order checked by reading, not run on hardware
//
#include <hip/hip_runtime.h>
#include <hip/hip_bf16.h>
#include <stdint.h>

// ---------------------------------------------------------------------------
// Problem constants (from reference)
// ---------------------------------------------------------------------------
#define ND   30000
#define NT   15000
#define FD   200
#define FT   570
#define HDIM 768
#define EDD  300000
#define EDT  400000
#define ETT  200000
#define NEG_SLOPE 0.2f
#define NTILES 48            // 768 / 16

typedef __attribute__((ext_vector_type(16))) __bf16 v16bf;
typedef __attribute__((ext_vector_type(8)))  float  v8f;

// ---------------------------------------------------------------------------
// bf16 helpers (round-to-nearest-even, manual bit twiddle: only used in the
// one-shot pack kernels, never in the GEMM hot loop)
// ---------------------------------------------------------------------------
__device__ __forceinline__ unsigned short f2bf(float f) {
    unsigned u = __float_as_uint(f);
    unsigned r = u + 0x7FFFu + ((u >> 16) & 1u);
    return (unsigned short)(r >> 16);
}
__device__ __forceinline__ float bf2f(unsigned short h) {
    return __uint_as_float(((unsigned)h) << 16);
}

// Fragment element k-offset map for V_WMMA_*_16X16X32 16-bit operands:
// lane<16 holds K {0..7,16..23}, lane>=16 holds K {8..15,24..31} of each chunk.
__device__ __forceinline__ int kmap(int lane, int i) {
    return ((lane >= 16) ? 8 : 0) + ((i < 8) ? i : i + 8);
}

// ---------------------------------------------------------------------------
// Pack A[M x K] (fp32, row-major) into WMMA fragment layout, bf16 hi/lo split.
// Fragment (rt, c) for lane L = 16 hi + 16 lo bf16 = 64 contiguous bytes at
// ushort index (((rt*CK + c)*32 + L) * 32).
// ---------------------------------------------------------------------------
__global__ void pack_a_kernel(const float* __restrict__ A, unsigned short* __restrict__ out,
                              int M, int K, int rtCount, int CK) {
    long i = (long)blockIdx.x * blockDim.x + threadIdx.x;
    long total = (long)rtCount * CK * 32 * 16;
    if (i >= total) return;
    int  i16  = (int)(i & 15);
    long t    = i >> 4;
    int  lane = (int)(t & 31);
    t >>= 5;
    int  c    = (int)(t % CK);
    long rt   = t / CK;
    int  row  = (int)(rt * 16 + (lane & 15));
    int  k    = c * 32 + kmap(lane, i16);
    float val = (row < M && k < K) ? A[(size_t)row * K + k] : 0.f;
    unsigned short h = f2bf(val);
    unsigned short l = f2bf(val - bf2f(h));
    size_t base = (((size_t)rt * CK + c) * 32 + lane) * 32;
    out[base + i16]      = h;
    out[base + 16 + i16] = l;
}

// ---------------------------------------------------------------------------
// Pack W[K x 768] (fp32, row-major) into B-fragment layout, bf16 hi/lo split.
// Fragment (c, nt) for lane L at ushort index (((c*48 + nt)*32 + L) * 32).
// ---------------------------------------------------------------------------
__global__ void pack_w_kernel(const float* __restrict__ W, unsigned short* __restrict__ out,
                              int K, int CK) {
    long i = (long)blockIdx.x * blockDim.x + threadIdx.x;
    long total = (long)CK * NTILES * 32 * 16;
    if (i >= total) return;
    int  i16  = (int)(i & 15);
    long t    = i >> 4;
    int  lane = (int)(t & 31);
    t >>= 5;
    int  nt   = (int)(t % NTILES);
    int  c    = (int)(t / NTILES);
    int  col  = nt * 16 + (lane & 15);
    int  k    = c * 32 + kmap(lane, i16);
    float val = (k < K) ? W[(size_t)k * HDIM + col] : 0.f;
    unsigned short h = f2bf(val);
    unsigned short l = f2bf(val - bf2f(h));
    size_t base = (((size_t)c * NTILES + nt) * 32 + lane) * 32;
    out[base + i16]      = h;
    out[base + 16 + i16] = l;
}

// ---------------------------------------------------------------------------
// GEMM on pre-packed fragments: C[M x 768] = A @ W, fp32-accurate via
// 3x bf16 WMMA (hi*hi + lo*hi + hi*lo). Block = 256 = 8 waves; wave owns one
// 16-row tile x 64 cols. Inner loop: 4+16 global_load_b128 + 12 WMMA.
// ---------------------------------------------------------------------------
union FragU { v16bf v; uint4 q[2]; };

__global__ __launch_bounds__(256)
void gemm_packed(const unsigned short* __restrict__ Apk,
                 const unsigned short* __restrict__ Wpk,
                 float* __restrict__ C, int M, int rtCount, int CK) {
    const int lane = threadIdx.x & 31;
    const int wave = threadIdx.x >> 5;
    const int rt   = blockIdx.x * 8 + wave;
    if (rt >= rtCount) return;               // wave-uniform, no barriers used
    const int nt0  = blockIdx.y * 4;

    v8f acc[4] = {};

    for (int c = 0; c < CK; ++c) {
        const uint4* ap = (const uint4*)(Apk + (((size_t)rt * CK + c) * 32 + lane) * 32);
        FragU aH, aL;
        aH.q[0] = ap[0]; aH.q[1] = ap[1];
        aL.q[0] = ap[2]; aL.q[1] = ap[3];
#pragma unroll
        for (int j = 0; j < 4; ++j) {
            const uint4* bp = (const uint4*)(Wpk + (((size_t)c * NTILES + nt0 + j) * 32 + lane) * 32);
            FragU bH, bL;
            bH.q[0] = bp[0]; bH.q[1] = bp[1];
            bL.q[0] = bp[2]; bL.q[1] = bp[3];
            acc[j] = __builtin_amdgcn_wmma_f32_16x16x32_bf16(false, aH.v, false, bH.v,
                                                             (short)0, acc[j], false, false);
            acc[j] = __builtin_amdgcn_wmma_f32_16x16x32_bf16(false, aL.v, false, bH.v,
                                                             (short)0, acc[j], false, false);
            acc[j] = __builtin_amdgcn_wmma_f32_16x16x32_bf16(false, aH.v, false, bL.v,
                                                             (short)0, acc[j], false, false);
        }
    }
    const int rbase = rt * 16 + ((lane >= 16) ? 8 : 0);
#pragma unroll
    for (int j = 0; j < 4; ++j) {
        int col = (nt0 + j) * 16 + (lane & 15);
#pragma unroll
        for (int v = 0; v < 8; ++v) {
            int r = rbase + v;
            if (r < M) C[(size_t)r * HDIM + col] = acc[j][v];
        }
    }
}

// ---------------------------------------------------------------------------
// out[row*4 + col] = dot(W[row, 0:768], a[0:768])   (attention vector precompute)
// ---------------------------------------------------------------------------
__global__ void rowdot_kernel(const float* __restrict__ W, const float* __restrict__ a,
                              float* __restrict__ out, int F, int col) {
    int lane = threadIdx.x & 31;
    int row  = blockIdx.x * 8 + (threadIdx.x >> 5);
    if (row >= F) return;
    float p = 0.f;
    for (int h = lane; h < HDIM; h += 32) p += W[(size_t)row * HDIM + h] * a[h];
#pragma unroll
    for (int m = 16; m > 0; m >>= 1) p += __shfl_xor(p, m, 32);
    if (lane == 0) out[row * 4 + col] = p;
}

// ---------------------------------------------------------------------------
// esed[node*4 + c] = dot(x[node, 0:F], V[0:F, c]), c = 0..3. one wave per node.
// ---------------------------------------------------------------------------
__global__ void gemv4_kernel(const float* __restrict__ x, const float* __restrict__ V,
                             float* __restrict__ esed, int N, int F) {
    int lane = threadIdx.x & 31;
    int node = blockIdx.x * 8 + (threadIdx.x >> 5);
    if (node >= N) return;
    float p0 = 0.f, p1 = 0.f, p2 = 0.f, p3 = 0.f;
    for (int k = lane; k < F; k += 32) {
        float xv = x[(size_t)node * F + k];
        p0 += xv * V[k * 4 + 0];
        p1 += xv * V[k * 4 + 1];
        p2 += xv * V[k * 4 + 2];
        p3 += xv * V[k * 4 + 3];
    }
#pragma unroll
    for (int m = 16; m > 0; m >>= 1) {
        p0 += __shfl_xor(p0, m, 32); p1 += __shfl_xor(p1, m, 32);
        p2 += __shfl_xor(p2, m, 32); p3 += __shfl_xor(p3, m, 32);
    }
    if (lane == 0) {
        esed[node * 4 + 0] = p0; esed[node * 4 + 1] = p1;
        esed[node * 4 + 2] = p2; esed[node * 4 + 3] = p3;
    }
}

// acc[i] = b1[i % 768] + b2[i % 768]
__global__ void initacc_kernel(float* __restrict__ acc, const float* __restrict__ b1,
                               const float* __restrict__ b2, long total) {
    long i = (long)blockIdx.x * blockDim.x + threadIdx.x;
    if (i >= total) return;
    int h = (int)(i % HDIM);
    acc[i] = b1[h] + b2[h];
}

// masked features: out[n,k] = mask[n] ? tok[k] : x[n,k]
__global__ void maskx_kernel(const float* __restrict__ x, const int* __restrict__ mask,
                             const float* __restrict__ tok, float* __restrict__ out,
                             long total, int F) {
    long i = (long)blockIdx.x * blockDim.x + threadIdx.x;
    if (i >= total) return;
    long n = i / F; int k = (int)(i % F);
    out[i] = mask[n] ? tok[k] : x[i];
}

// ---------------------------------------------------------------------------
// order-preserving float <-> unsigned encoding for atomicMax
// ---------------------------------------------------------------------------
__device__ __forceinline__ unsigned fenc(float f) {
    unsigned u = __float_as_uint(f);
    return (u & 0x80000000u) ? ~u : (u | 0x80000000u);
}
__device__ __forceinline__ float fdec(unsigned e) {
    unsigned u = (e & 0x80000000u) ? (e & 0x7FFFFFFFu) : ~e;
    return __uint_as_float(u);
}
__device__ __forceinline__ float leaky(float v) { return v > 0.f ? v : NEG_SLOPE * v; }

// pass 1: e = leaky_relu(es[src]+ed[dst]); wbuf[e]=e; segment max via atomicMax
__global__ void edge_max_kernel(const int* __restrict__ S, const int* __restrict__ D,
                                int Ereal, int nSelf,
                                const float* __restrict__ esA, int cs,
                                const float* __restrict__ edA, int cd,
                                float* __restrict__ wbuf, unsigned* __restrict__ menc) {
    int e = blockIdx.x * blockDim.x + threadIdx.x;
    int Etot = Ereal + nSelf;
    if (e >= Etot) return;
    int s, d;
    if (e < Ereal) { s = S[e]; d = D[e]; } else { s = d = e - Ereal; }
    float v = leaky(esA[s * 4 + cs] + edA[d * 4 + cd]);
    wbuf[e] = v;
    atomicMax(&menc[d], fenc(v));
}

// pass 2: w = exp(e - m[dst]); wbuf[e]=w; segment sum via atomicAdd
__global__ void edge_exp_kernel(const int* __restrict__ D, int Ereal, int nSelf,
                                float* __restrict__ wbuf, const unsigned* __restrict__ menc,
                                float* __restrict__ ssum) {
    int e = blockIdx.x * blockDim.x + threadIdx.x;
    int Etot = Ereal + nSelf;
    if (e >= Etot) return;
    int d = (e < Ereal) ? D[e] : e - Ereal;
    unsigned me = menc[d];
    float m = (me == 0u) ? 0.f : fdec(me);
    float w = __expf(wbuf[e] - m);
    wbuf[e] = w;
    atomicAdd(&ssum[d], w);
}

// pass 3: acc[dst, :] += hs[src, :] * (w / (s[dst] + 1e-16)); 4 edges per block
__global__ void edge_agg_kernel(const int* __restrict__ S, const int* __restrict__ D,
                                int Ereal, int nSelf,
                                const float* __restrict__ hs, const float* __restrict__ wbuf,
                                const float* __restrict__ ssum, float* __restrict__ acc) {
    int Etot = Ereal + nSelf;
#pragma unroll
    for (int eo = 0; eo < 4; ++eo) {
        int e = blockIdx.x * 4 + eo;
        if (e >= Etot) continue;
        int s, d;
        if (e < Ereal) { s = S[e]; d = D[e]; } else { s = d = e - Ereal; }
        float alpha = wbuf[e] / (ssum[d] + 1e-16f);
        const float* hrow = hs + (size_t)s * HDIM;
        float*       arow = acc + (size_t)d * HDIM;
        for (int h = threadIdx.x; h < HDIM; h += blockDim.x)
            atomicAdd(&arow[h], hrow[h] * alpha);
    }
}

__global__ void relu_store_kernel(const float* __restrict__ acc, float* __restrict__ out,
                                  long total) {
    long i = (long)blockIdx.x * blockDim.x + threadIdx.x;
    if (i >= total) return;
    out[i] = fmaxf(acc[i], 0.f);
}

__global__ void count_mask_kernel(const int* __restrict__ mask, int N, int* __restrict__ cnt) {
    int n = blockIdx.x * blockDim.x + threadIdx.x;
    if (n < N && mask[n]) atomicAdd(cnt, 1);
}

// sum over masked nodes of (out1 - relu(acc2))^2 ; wave-reduced atomics
__global__ void loss_acc_kernel(const float* __restrict__ out1, const float* __restrict__ acc2,
                                const int* __restrict__ mask, long total,
                                float* __restrict__ num) {
    long i = (long)blockIdx.x * blockDim.x + threadIdx.x;
    float local = 0.f;
    if (i < total) {
        long n = i / HDIM;
        if (mask[n]) {
            float diff = out1[i] - fmaxf(acc2[i], 0.f);
            local = diff * diff;
        }
    }
#pragma unroll
    for (int m = 16; m > 0; m >>= 1) local += __shfl_xor(local, m, 32);
    if ((threadIdx.x & 31) == 0 && local != 0.f) atomicAdd(num, local);
}

__global__ void finalize_kernel(const float* __restrict__ numd, const float* __restrict__ numt,
                                const int* __restrict__ cntd, const int* __restrict__ cntt,
                                float* __restrict__ out) {
    if (threadIdx.x == 0 && blockIdx.x == 0) {
        float l = numd[0] / ((float)cntd[0] * (float)HDIM + 1e-16f)
                + numt[0] / ((float)cntt[0] * (float)HDIM + 1e-16f);
        out[0] = l;
    }
}

// ---------------------------------------------------------------------------
// host-side orchestration
// ---------------------------------------------------------------------------
static inline size_t alignUp(size_t x) { return (x + 255) & ~(size_t)255; }

extern "C" void kernel_launch(void* const* d_in, const int* in_sizes, int n_in,
                              void* d_out, int out_size, void* d_ws, size_t ws_size,
                              hipStream_t stream) {
    (void)in_sizes; (void)n_in; (void)out_size; (void)ws_size;

    const float* drug_x   = (const float*)d_in[0];
    const float* target_x = (const float*)d_in[1];
    const int*   dd_src   = (const int*)d_in[2];
    const int*   dd_dst   = (const int*)d_in[3];
    const int*   dt_src   = (const int*)d_in[4];
    const int*   dt_dst   = (const int*)d_in[5];
    const int*   tt_src   = (const int*)d_in[6];
    const int*   tt_dst   = (const int*)d_in[7];
    const int*   drug_mask   = (const int*)d_in[8];
    const int*   target_mask = (const int*)d_in[9];
    const float* mask_drug   = (const float*)d_in[10];
    const float* mask_target = (const float*)d_in[11];
    const float* Wsrc_dd = (const float*)d_in[12]; const float* Wdst_dd = (const float*)d_in[13];
    const float* asrc_dd = (const float*)d_in[14]; const float* adst_dd = (const float*)d_in[15];
    const float* b_dd    = (const float*)d_in[16];
    const float* Wsrc_dt = (const float*)d_in[17]; const float* Wdst_dt = (const float*)d_in[18];
    const float* asrc_dt = (const float*)d_in[19]; const float* adst_dt = (const float*)d_in[20];
    const float* b_dt    = (const float*)d_in[21];
    const float* Wsrc_rv = (const float*)d_in[22]; const float* Wdst_rv = (const float*)d_in[23];
    const float* asrc_rv = (const float*)d_in[24]; const float* adst_rv = (const float*)d_in[25];
    const float* b_rv    = (const float*)d_in[26];
    const float* Wsrc_tt = (const float*)d_in[27]; const float* Wdst_tt = (const float*)d_in[28];
    const float* asrc_tt = (const float*)d_in[29]; const float* adst_tt = (const float*)d_in[30];
    const float* b_tt    = (const float*)d_in[31];

    // fragment-chunk counts
    const int CKD = (FD + 31) / 32;          // 7
    const int CKT = (FT + 31) / 32;          // 18
    const int RTD = (ND + 15) / 16;          // 1875
    const int RTT = (NT + 15) / 16;          // 938

    // ---- workspace carve-up ----
    char* ws = (char*)d_ws;
    size_t off = 0;
    float*    hs    = (float*)(ws + off); off += alignUp((size_t)ND * HDIM * 4);
    float*    accd  = (float*)(ws + off); off += alignUp((size_t)ND * HDIM * 4);
    float*    acct  = (float*)(ws + off); off += alignUp((size_t)NT * HDIM * 4);
    float*    mxd   = (float*)(ws + off); off += alignUp((size_t)ND * FD * 4);
    float*    mxt   = (float*)(ws + off); off += alignUp((size_t)NT * FT * 4);
    // shared A-pack buffer, sized for the larger of the two node types
    size_t apkD = (size_t)RTD * CKD * 32 * 32;     // ushort count, drug
    size_t apkT = (size_t)RTT * CKT * 32 * 32;     // ushort count, target
    unsigned short* apack = (unsigned short*)(ws + off);
    off += alignUp((apkD > apkT ? apkD : apkT) * 2);
    // packed weights (persist across both passes)
    size_t wpkD = (size_t)CKD * NTILES * 32 * 32;  // ushort count, K=FD
    size_t wpkT = (size_t)CKT * NTILES * 32 * 32;  // ushort count, K=FT
    unsigned short* wpk_dd = (unsigned short*)(ws + off); off += alignUp(wpkD * 2);
    unsigned short* wpk_dt = (unsigned short*)(ws + off); off += alignUp(wpkD * 2);
    unsigned short* wpk_rv = (unsigned short*)(ws + off); off += alignUp(wpkT * 2);
    unsigned short* wpk_tt = (unsigned short*)(ws + off); off += alignUp(wpkT * 2);
    float*    esedd = (float*)(ws + off); off += alignUp((size_t)ND * 4 * 4);
    float*    esedt = (float*)(ws + off); off += alignUp((size_t)NT * 4 * 4);
    float*    wbuf  = (float*)(ws + off); off += alignUp((size_t)EDT * 4);
    unsigned* menc  = (unsigned*)(ws + off); off += alignUp((size_t)ND * 4);
    float*    ssum  = (float*)(ws + off); off += alignUp((size_t)ND * 4);
    float*    Vd    = (float*)(ws + off); off += alignUp((size_t)FD * 4 * 4);
    float*    Vt    = (float*)(ws + off); off += alignUp((size_t)FT * 4 * 4);
    float*    numd  = (float*)(ws + off);
    float*    numt  = numd + 1;
    int*      cntd  = (int*)(numd + 2);
    int*      cntt  = cntd + 1;

    float* out_d = (float*)d_out;
    float* out_t = out_d + (size_t)ND * HDIM;
    float* out_l = out_t + (size_t)NT * HDIM;

    // ---- pack weights once ----
    {
        long tD = (long)CKD * NTILES * 32 * 16, tT = (long)CKT * NTILES * 32 * 16;
        pack_w_kernel<<<(unsigned)((tD + 255) / 256), 256, 0, stream>>>(Wsrc_dd, wpk_dd, FD, CKD);
        pack_w_kernel<<<(unsigned)((tD + 255) / 256), 256, 0, stream>>>(Wsrc_dt, wpk_dt, FD, CKD);
        pack_w_kernel<<<(unsigned)((tT + 255) / 256), 256, 0, stream>>>(Wsrc_rv, wpk_rv, FT, CKT);
        pack_w_kernel<<<(unsigned)((tT + 255) / 256), 256, 0, stream>>>(Wsrc_tt, wpk_tt, FT, CKT);
    }

    // ---- precompute fused attention vectors (W @ a) ----
    // Vd cols: 0 es_dd, 1 ed_dd, 2 ed_rv, 3 es_dt ; Vt cols: 0 es_rv, 1 ed_dt, 2 es_tt, 3 ed_tt
    rowdot_kernel<<<(FD + 7) / 8, 256, 0, stream>>>(Wsrc_dd, asrc_dd, Vd, FD, 0);
    rowdot_kernel<<<(FD + 7) / 8, 256, 0, stream>>>(Wdst_dd, adst_dd, Vd, FD, 1);
    rowdot_kernel<<<(FD + 7) / 8, 256, 0, stream>>>(Wdst_rv, adst_rv, Vd, FD, 2);
    rowdot_kernel<<<(FD + 7) / 8, 256, 0, stream>>>(Wsrc_dt, asrc_dt, Vd, FD, 3);
    rowdot_kernel<<<(FT + 7) / 8, 256, 0, stream>>>(Wsrc_rv, asrc_rv, Vt, FT, 0);
    rowdot_kernel<<<(FT + 7) / 8, 256, 0, stream>>>(Wdst_dt, adst_dt, Vt, FT, 1);
    rowdot_kernel<<<(FT + 7) / 8, 256, 0, stream>>>(Wsrc_tt, asrc_tt, Vt, FT, 2);
    rowdot_kernel<<<(FT + 7) / 8, 256, 0, stream>>>(Wdst_tt, adst_tt, Vt, FT, 3);

    // one GAT relation: pack A, WMMA GEMM to hs, segment softmax, scatter to acc
    auto relation = [&](const float* xsrc, int Msrc, int Ksrc, int CK, int rtCount,
                        const unsigned short* wpk,
                        const int* S, const int* D, int Ereal, int nSelf, int nDst,
                        const float* esA, int cs, const float* edA, int cd, float* acc) {
        long tp = (long)rtCount * CK * 32 * 16;
        pack_a_kernel<<<(unsigned)((tp + 255) / 256), 256, 0, stream>>>(xsrc, apack,
                                                                        Msrc, Ksrc, rtCount, CK);
        dim3 gg((rtCount + 7) / 8, HDIM / 64);
        gemm_packed<<<gg, 256, 0, stream>>>(apack, wpk, hs, Msrc, rtCount, CK);
        hipMemsetAsync(menc, 0, (size_t)nDst * 4, stream);
        hipMemsetAsync(ssum, 0, (size_t)nDst * 4, stream);
        int Etot = Ereal + nSelf;
        edge_max_kernel<<<(Etot + 255) / 256, 256, 0, stream>>>(S, D, Ereal, nSelf,
                                                                esA, cs, edA, cd, wbuf, menc);
        edge_exp_kernel<<<(Etot + 255) / 256, 256, 0, stream>>>(D, Ereal, nSelf, wbuf, menc, ssum);
        edge_agg_kernel<<<(Etot + 3) / 4, 256, 0, stream>>>(S, D, Ereal, nSelf,
                                                            hs, wbuf, ssum, acc);
    };

    auto layer = [&](const float* xd, const float* xt) {
        gemv4_kernel<<<(ND + 7) / 8, 256, 0, stream>>>(xd, Vd, esedd, ND, FD);
        gemv4_kernel<<<(NT + 7) / 8, 256, 0, stream>>>(xt, Vt, esedt, NT, FT);
        long td = (long)ND * HDIM, ttl = (long)NT * HDIM;
        initacc_kernel<<<(unsigned)((td + 255) / 256), 256, 0, stream>>>(accd, b_dd, b_rv, td);
        initacc_kernel<<<(unsigned)((ttl + 255) / 256), 256, 0, stream>>>(acct, b_dt, b_tt, ttl);
        // d->d (self loops): src es = d col0, dst ed = d col1
        relation(xd, ND, FD, CKD, RTD, wpk_dd, dd_src, dd_dst, EDD, ND, ND,
                 esedd, 0, esedd, 1, accd);
        // t->d reverse: src=target (es_rv = t col0), dst=drug (ed_rv = d col2)
        relation(xt, NT, FT, CKT, RTT, wpk_rv, dt_dst, dt_src, EDT, 0, ND,
                 esedt, 0, esedd, 2, accd);
        // d->t: src=drug (es_dt = d col3), dst=target (ed_dt = t col1)
        relation(xd, ND, FD, CKD, RTD, wpk_dt, dt_src, dt_dst, EDT, 0, NT,
                 esedd, 3, esedt, 1, acct);
        // t->t (self loops): src es = t col2, dst ed = t col3
        relation(xt, NT, FT, CKT, RTT, wpk_tt, tt_src, tt_dst, ETT, NT, NT,
                 esedt, 2, esedt, 3, acct);
    };

    // ---- pass 1: clean features -> d_out ----
    layer(drug_x, target_x);
    long td = (long)ND * HDIM, ttl = (long)NT * HDIM;
    relu_store_kernel<<<(unsigned)((td + 255) / 256), 256, 0, stream>>>(accd, out_d, td);
    relu_store_kernel<<<(unsigned)((ttl + 255) / 256), 256, 0, stream>>>(acct, out_t, ttl);

    // ---- pass 2: masked features -> acc buffers, then MSE loss ----
    long md = (long)ND * FD, mt = (long)NT * FT;
    maskx_kernel<<<(unsigned)((md + 255) / 256), 256, 0, stream>>>(drug_x, drug_mask,
                                                                   mask_drug, mxd, md, FD);
    maskx_kernel<<<(unsigned)((mt + 255) / 256), 256, 0, stream>>>(target_x, target_mask,
                                                                   mask_target, mxt, mt, FT);
    layer(mxd, mxt);

    hipMemsetAsync(numd, 0, 4 * sizeof(float), stream);
    count_mask_kernel<<<(ND + 255) / 256, 256, 0, stream>>>(drug_mask, ND, cntd);
    count_mask_kernel<<<(NT + 255) / 256, 256, 0, stream>>>(target_mask, NT, cntt);
    loss_acc_kernel<<<(unsigned)((td + 255) / 256), 256, 0, stream>>>(out_d, accd, drug_mask,
                                                                      td, numd);
    loss_acc_kernel<<<(unsigned)((ttl + 255) / 256), 256, 0, stream>>>(out_t, acct, target_mask,
                                                                       ttl, numt);
    finalize_kernel<<<1, 1, 0, stream>>>(numd, numt, cntd, cntt, out_l);
}
